// SingleStageDetectorAdv_807453852174
// MI455X (gfx1250) — compile-verified, matrix operand order checked by reading, not run on hardware
//
#include <hip/hip_runtime.h>
#include <hip/hip_bf16.h>
#include <stdint.h>

// ---------------- problem constants (match reference) ----------------
#define NUM_CLASSES 80
#define NUM_ANCH    9
#define BATCH       4
#define NLEVEL      5
#define NPAIR       (BATCH * NLEVEL)
#define NBINS       1024
#define NREP        4                     // LDS histogram replicas (conflict relief)
#define NMS_PRE     1000
#define CAND        (NLEVEL * NMS_PRE)    // 5000 per image
#define MAX_DET     100
#define SCORE_THR   0.05f
#define LOGIT_THR   (-2.9444389791664403f)   // log(0.05/0.95): sigmoid(x)>0.05 <=> x>this
#define IOU_THR     0.5f
#define IMG_W       1024.0f
#define IMG_H       1024.0f
#define MAX_RATIO   4.135166556742356f       // ln(1000/16)

// ---------------- workspace layout (bytes) ----------------
#define WS_HIST   0                              // NPAIR*NBINS u32      = 81920
#define WS_CUT    81920                          // NPAIR int            = 80
#define WS_CNT    82000                          // NPAIR int            = 80
#define WS_BOX    82176                          // BATCH*CAND*4 f32     = 320000
#define WS_OBOX   402176                         // BATCH*CAND*4 f32     = 320000
#define WS_AREA   722176                         // BATCH*CAND f32       = 80000
#define WS_SCORE  802176                         // BATCH*CAND f32       = 80000
#define WS_LABEL  882176                         // BATCH*CAND int       = 80000

typedef __attribute__((ext_vector_type(16))) _Float16 v16h;
typedef __attribute__((ext_vector_type(8)))  float    v8f;

// orderable key: monotone map f32 -> u32 (larger key == larger float)
__device__ __forceinline__ unsigned order_key(float f) {
    unsigned u = __float_as_uint(f);
    return ((int)u < 0) ? ~u : (u | 0x80000000u);
}

// ---------------- init: zero histograms, poison candidate slots ----------------
__global__ void det_init_kernel(unsigned* __restrict__ hist, float* __restrict__ score,
                                int* __restrict__ label, float* __restrict__ area,
                                float* __restrict__ box, float* __restrict__ obox) {
    int t = blockIdx.x * blockDim.x + threadIdx.x;
    if (t < NPAIR * NBINS) hist[t] = 0u;
    if (t < BATCH * CAND) {
        score[t] = -1.0f;
        label[t] = -1;
        area[t]  = 0.0f;
        for (int k = 0; k < 4; ++k) { box[t * 4 + k] = 0.0f; obox[t * 4 + k] = 0.0f; }
    }
}

// ---------------- pass 1: 1024-bin histogram of logit keys (streaming, f32x4) ----
// Logits are ~N(0,1): the top-10-bit key concentrates into a few dozen hot bins,
// so the LDS histogram is replicated NREP-wide (lane&3) to cut ds_add conflicts.
__global__ __launch_bounds__(256) void det_hist_kernel(const float* __restrict__ cls,
                                                       unsigned* __restrict__ hist,
                                                       int level, int perImg4) {
    __shared__ unsigned lh[NREP][NBINS];     // 16 KB; still tiny vs 320 KB/WGP
    for (int i = threadIdx.x; i < NREP * NBINS; i += blockDim.x)
        lh[i / NBINS][i % NBINS] = 0u;
    __syncthreads();

    const int img = blockIdx.y;
    const int rep = threadIdx.x & (NREP - 1);
    const long long e4 = (long long)blockIdx.x * blockDim.x + threadIdx.x;
    if (e4 < perImg4) {
        const float4* p4 = (const float4*)cls + (long long)img * perImg4;
        __builtin_prefetch((const char*)(p4 + e4) + 16384, 0, 3);   // global_prefetch_b8
        float4 v = p4[e4];
        float vv[4] = {v.x, v.y, v.z, v.w};
#pragma unroll
        for (int k = 0; k < 4; ++k) {
            if (vv[k] > LOGIT_THR) {
                unsigned bin = order_key(vv[k]) >> 22;   // top 10 bits
                atomicAdd(&lh[rep][bin], 1u);
            }
        }
    }
    __syncthreads();
    unsigned* gh = hist + (size_t)(img * NLEVEL + level) * NBINS;
    for (int i = threadIdx.x; i < NBINS; i += blockDim.x) {
        unsigned s = lh[0][i] + lh[1][i] + lh[2][i] + lh[3][i];
        if (s) atomicAdd(&gh[i], s);
    }
}

// ---------------- pass 2: cutoff bin per (img,level) ----------------------------
// WMMA coarse reduction: A(16x32 f16) x ones(32x16) -> D[m][*] = sum of 32 bins of
// tile m. Two WMMAs cover all 1024 bins -> 32 tile sums. K-permutation inside a
// row is irrelevant since B is all-ones. Then lane 0 scans tiles coarse-to-fine.
__global__ void det_cutoff_kernel(const unsigned* __restrict__ hist,
                                  int* __restrict__ cutBin, int* __restrict__ candCount) {
    const int pair = blockIdx.x;
    const int lane = threadIdx.x;          // 32 threads, EXEC all ones at WMMA
    const unsigned* h = hist + (size_t)pair * NBINS;
    __shared__ float coarse[32];

    const int row = lane & 15, half = lane >> 4;
    v16h a0, a1, ones;
    v8f  c0 = {}, c1 = {};
#pragma unroll
    for (int j = 0; j < 16; ++j) {
        ones[j] = (_Float16)1.0f;
        a0[j] = (_Float16)(float)h[row * 32 + half * 16 + j];          // tiles 0..15
        a1[j] = (_Float16)(float)h[512 + row * 32 + half * 16 + j];    // tiles 16..31
    }
    v8f d0 = __builtin_amdgcn_wmma_f32_16x16x32_f16(false, a0, false, ones, (short)0, c0, false, false);
    v8f d1 = __builtin_amdgcn_wmma_f32_16x16x32_f16(false, a1, false, ones, (short)0, c1, false, false);

    // D layout: lane 0 col 0 holds rows 0..7 in its 8 VGPRs; lane 16 rows 8..15.
    if (lane == 0)  { for (int r = 0; r < 8; ++r) { coarse[r]      = d0[r]; coarse[16 + r] = d1[r]; } }
    if (lane == 16) { for (int r = 0; r < 8; ++r) { coarse[8 + r]  = d0[r]; coarse[24 + r] = d1[r]; } }
    __syncthreads();

    if (lane == 0) {
        // walk tiles from the highest keys (largest scores) down
        float acc = 0.0f; int tile = 31;
        for (; tile >= 0; --tile) {
            if (acc + coarse[tile] >= (float)NMS_PRE) break;
            acc += coarse[tile];
        }
        int cut = 0;
        if (tile >= 0) {
            unsigned uacc = (unsigned)(acc + 0.5f);
            cut = tile * 32;
            for (int b = tile * 32 + 31; b >= tile * 32; --b) {
                uacc += h[b];
                if (uacc >= NMS_PRE) { cut = b; break; }
            }
        }
        cutBin[pair] = cut;
        candCount[pair] = 0;
    }
}

// ---------------- pass 3: compact winners + anchor decode ------------------------
__global__ __launch_bounds__(256) void det_compact_kernel(
        const float* __restrict__ cls, const float* __restrict__ bbox,
        const int* __restrict__ cutBin, int* __restrict__ candCount,
        float* __restrict__ boxO, float* __restrict__ oboxO, float* __restrict__ areaO,
        float* __restrict__ scoreO, int* __restrict__ labelO,
        int level, int H, int W, float stride, int N /* = 720*H*W */) {
    const int img = blockIdx.y;
    const int e = blockIdx.x * blockDim.x + threadIdx.x;
    if (e >= N) return;

    const float x = cls[(long long)img * N + e];
    if (x <= LOGIT_THR) return;
    const int pair = img * NLEVEL + level;
    const unsigned bin = order_key(x) >> 22;
    if ((int)bin < cutBin[pair]) return;

    const int slot = atomicAdd(&candCount[pair], 1);
    if (slot >= NMS_PRE) return;

    // element -> (anchor a, class c, h, w); memory layout is [A*C, H, W]
    const int hw = H * W;
    const int ch = e / hw;
    const int rem = e - ch * hw;
    const int hh = rem / W, ww = rem - (rem / W) * W;
    const int a = ch / NUM_CLASSES, c = ch - a * NUM_CLASSES;

    // RetinaNet base anchors: base=4*stride, ratios (0.5,1,2), 3 scales/octave
    const float hr_tab[3] = {0.70710678118654752f, 1.0f, 1.41421356237309505f};
    const int ri = a / 3, si = a - ri * 3;
    const float scl = exp2f((float)si * (1.0f / 3.0f));
    const float base = 4.0f * stride;
    const float ah = base * hr_tab[ri] * scl;
    const float aw = base * (1.0f / hr_tab[ri]) * scl;
    const float px = (float)ww * stride, py = (float)hh * stride;

    // deltas from bbox[img, a*4+d, h, w]
    const long long bb = ((long long)img * (NUM_ANCH * 4) + a * 4) * hw + (long long)hh * W + ww;
    const float dx = bbox[bb];
    const float dy = bbox[bb + (long long)hw];
    float dw = bbox[bb + 2LL * hw];
    float dh = bbox[bb + 3LL * hw];
    dw = fminf(fmaxf(dw, -MAX_RATIO), MAX_RATIO);
    dh = fminf(fmaxf(dh, -MAX_RATIO), MAX_RATIO);

    const float gx = px + aw * dx, gy = py + ah * dy;
    const float gw = aw * __expf(dw), gh = ah * __expf(dh);
    const float x1 = fminf(fmaxf(gx - 0.5f * gw, 0.0f), IMG_W);
    const float y1 = fminf(fmaxf(gy - 0.5f * gh, 0.0f), IMG_H);
    const float x2 = fminf(fmaxf(gx + 0.5f * gw, 0.0f), IMG_W);
    const float y2 = fminf(fmaxf(gy + 0.5f * gh, 0.0f), IMG_H);

    const float s = 1.0f / (1.0f + __expf(-x));
    const float off = (float)c * (IMG_W + 1.0f);

    const int gi = img * CAND + level * NMS_PRE + slot;
    boxO[gi * 4 + 0] = x1;  boxO[gi * 4 + 1] = y1;
    boxO[gi * 4 + 2] = x2;  boxO[gi * 4 + 3] = y2;
    oboxO[gi * 4 + 0] = x1 + off;  oboxO[gi * 4 + 1] = y1 + off;
    oboxO[gi * 4 + 2] = x2 + off;  oboxO[gi * 4 + 3] = y2 + off;
    areaO[gi]  = (x2 - x1) * (y2 - y1);
    scoreO[gi] = s;
    labelO[gi] = c;
}

// ---------------- pass 4: greedy class-aware NMS, 1 block / image ----------------
__global__ __launch_bounds__(1024) void det_nms_kernel(
        const float* __restrict__ boxes, const float* __restrict__ oboxes,
        const float* __restrict__ areas, const float* __restrict__ scoresIn,
        const int* __restrict__ labels, float* __restrict__ out) {
    const int img = blockIdx.x, tid = threadIdx.x;
    __shared__ float sScore[CAND];
    __shared__ float wVal[32];
    __shared__ int   wIdx[32];
    __shared__ int   sBest;

    // Stage scores into LDS with the CDNA5 async global->LDS path (ASYNCcnt).
    {
        const float* gS = scoresIn + (size_t)img * CAND;
        const unsigned long long sa = (unsigned long long)(uintptr_t)gS;
        for (int j = tid; j < CAND; j += 1024) {
            unsigned lds = (unsigned)(uintptr_t)&sScore[j];   // low 32 bits = LDS offset
            unsigned voff = (unsigned)(j * 4);
            asm volatile("global_load_async_to_lds_b32 %0, %1, %2 offset:0"
                         :: "v"(lds), "v"(voff), "s"(sa) : "memory");
        }
        asm volatile("s_wait_asynccnt 0" ::: "memory");
    }
    __syncthreads();

    const float* gB = boxes  + (size_t)img * CAND * 4;
    const float* gO = oboxes + (size_t)img * CAND * 4;
    const float* gA = areas  + (size_t)img * CAND;
    const int*   gL = labels + (size_t)img * CAND;
    const int lane = tid & 31, wave = tid >> 5;

    for (int it = 0; it < MAX_DET; ++it) {
        // --- argmax over sScore[CAND]: per-thread scan + wave32 shuffle + LDS merge
        float bv = -2.0f; int bi = 0;
        for (int j = tid; j < CAND; j += 1024) {
            float s = sScore[j];
            if (s > bv) { bv = s; bi = j; }
        }
#pragma unroll
        for (int m = 16; m >= 1; m >>= 1) {
            float ov = __shfl_xor(bv, m);
            int   oi = __shfl_xor(bi, m);
            if (ov > bv) { bv = ov; bi = oi; }
        }
        if (lane == 0) { wVal[wave] = bv; wIdx[wave] = bi; }
        __syncthreads();
        if (tid == 0) {
            float v = wVal[0]; int ix = wIdx[0];
            for (int k = 1; k < 32; ++k) if (wVal[k] > v) { v = wVal[k]; ix = wIdx[k]; }
            sBest = ix;
            const bool valid = v > SCORE_THR;
            float* drow = out + (size_t)(img * MAX_DET + it) * 5;
            drow[0] = valid ? gB[ix * 4 + 0] : 0.0f;
            drow[1] = valid ? gB[ix * 4 + 1] : 0.0f;
            drow[2] = valid ? gB[ix * 4 + 2] : 0.0f;
            drow[3] = valid ? gB[ix * 4 + 3] : 0.0f;
            drow[4] = valid ? v : 0.0f;
            out[(size_t)BATCH * MAX_DET * 5 + img * MAX_DET + it] = valid ? (float)gL[ix] : -1.0f;
        }
        __syncthreads();

        // --- suppression vs best (class-offset boxes; strict iou > 0.5, kills self)
        const int b = sBest;
        const float bx1 = gO[b * 4 + 0], by1 = gO[b * 4 + 1];
        const float bx2 = gO[b * 4 + 2], by2 = gO[b * 4 + 3];
        const float bA  = gA[b];
        for (int j = tid; j < CAND; j += 1024) {
            const float x1 = fmaxf(bx1, gO[j * 4 + 0]);
            const float y1 = fmaxf(by1, gO[j * 4 + 1]);
            const float x2 = fminf(bx2, gO[j * 4 + 2]);
            const float y2 = fminf(by2, gO[j * 4 + 3]);
            const float inter = fmaxf(x2 - x1, 0.0f) * fmaxf(y2 - y1, 0.0f);
            const float iou = inter / (gA[j] + bA - inter + 1e-6f);
            if (iou > IOU_THR) sScore[j] = -1.0f;
        }
        __syncthreads();
    }
}

// ---------------- host launch ----------------
extern "C" void kernel_launch(void* const* d_in, const int* in_sizes, int n_in,
                              void* d_out, int out_size, void* d_ws, size_t ws_size,
                              hipStream_t stream) {
    (void)in_sizes; (void)n_in; (void)out_size; (void)ws_size;
    static const int   Hs[NLEVEL] = {128, 64, 32, 16, 8};
    static const int   Ws[NLEVEL] = {128, 64, 32, 16, 8};
    static const float St[NLEVEL] = {8.0f, 16.0f, 32.0f, 64.0f, 128.0f};

    uint8_t* ws = (uint8_t*)d_ws;
    unsigned* hist  = (unsigned*)(ws + WS_HIST);
    int*      cut   = (int*)(ws + WS_CUT);
    int*      cnt   = (int*)(ws + WS_CNT);
    float*    box   = (float*)(ws + WS_BOX);
    float*    obox  = (float*)(ws + WS_OBOX);
    float*    area  = (float*)(ws + WS_AREA);
    float*    score = (float*)(ws + WS_SCORE);
    int*      label = (int*)(ws + WS_LABEL);

    // 1) init
    {
        int tot = NPAIR * NBINS;                       // 20480 >= 20000
        det_init_kernel<<<(tot + 255) / 256, 256, 0, stream>>>(hist, score, label, area, box, obox);
    }
    // 2) histogram pass per level
    for (int l = 0; l < NLEVEL; ++l) {
        const float* cls = (const float*)d_in[l];
        int N = NUM_ANCH * NUM_CLASSES * Hs[l] * Ws[l];
        int perImg4 = N / 4;
        dim3 grid((perImg4 + 255) / 256, BATCH);
        det_hist_kernel<<<grid, 256, 0, stream>>>(cls, hist, l, perImg4);
    }
    // 3) cutoff select (one wave per (img,level); WMMA coarse tile sums)
    det_cutoff_kernel<<<NPAIR, 32, 0, stream>>>(hist, cut, cnt);
    // 4) compaction + decode per level
    for (int l = 0; l < NLEVEL; ++l) {
        const float* cls  = (const float*)d_in[l];
        const float* bbox = (const float*)d_in[NLEVEL + l];
        int N = NUM_ANCH * NUM_CLASSES * Hs[l] * Ws[l];
        dim3 grid((N + 255) / 256, BATCH);
        det_compact_kernel<<<grid, 256, 0, stream>>>(cls, bbox, cut, cnt,
                                                     box, obox, area, score, label,
                                                     l, Hs[l], Ws[l], St[l], N);
    }
    // 5) NMS, one block per image
    det_nms_kernel<<<BATCH, 1024, 0, stream>>>(box, obox, area, score, label, (float*)d_out);
}
